// Attention_3238405341880
// MI455X (gfx1250) — matrix operand order repaired and checked by reading, not compiled, not run on hardware
//
#include <hip/hip_runtime.h>

#define BATCH  4
#define SEQ    4096
#define EDIM   256
#define DDIM   64
#define KSPLIT 4

typedef __attribute__((ext_vector_type(16))) __bf16        v16bf;
typedef __attribute__((ext_vector_type(8)))  float         v8f;
typedef __attribute__((ext_vector_type(4)))  float         v4f;
typedef __attribute__((ext_vector_type(4)))  unsigned int  v4u;

union Frag {
  unsigned short u[16];
  __bf16         b[16];
  v4u            q[2];
  v16bf          v;
};

__device__ __forceinline__ unsigned short f2bf(float f) {
  union { float f; unsigned int u; } x; x.f = f;
  unsigned int u = x.u;
  u = u + 0x7fffu + ((u >> 16) & 1u);   // round-to-nearest-even
  return (unsigned short)(u >> 16);
}

// A-matrix (16x32, 16-bit) element -> K index, per ISA 7.12.2
// For fixed half, elements 0..7 map to K = 8h..8h+7 and 8..15 to K = 16+8h..23+8h.
__device__ __forceinline__ int aidx_of(int e, int half) {
  int p = e >> 1;
  return (p & 3) * 2 + (p >> 2) * 16 + half * 8 + (e & 1);
}

// ---------------------------------------------------------------------------
// One-time W -> bf16 conversion: wb = [3][DDIM][EDIM] bf16 (q,k,v order)
// ---------------------------------------------------------------------------
__global__ void __launch_bounds__(256)
wcvt_kernel(const float* __restrict__ Wq, const float* __restrict__ Wk,
            const float* __restrict__ Wv, unsigned int* __restrict__ wb) {
  const int t = blockIdx.x * 256 + threadIdx.x;       // pair index
  const int per = (DDIM * EDIM) / 2;                  // 8192 pairs per matrix
  const int w = t / per;
  const int i = t % per;
  const float* W = (w == 0) ? Wq : (w == 1) ? Wk : Wv;
  unsigned int lo = f2bf(W[2 * i]);
  unsigned int hi = f2bf(W[2 * i + 1]);
  wb[t] = lo | (hi << 16);
}

// ---------------------------------------------------------------------------
// WMMA projection: one wave = 16 tokens x 64 outputs, K = 256 (8 chunks).
//   p==0: qp[b,n,d] = bf16( (X W^T) * 0.125 * log2(e) )   (scale folded for exp2)
//   p==1: kp[b,n,d] = bf16(  X W^T )
//   p==2: vt[b,d,n] = bf16(  X W^T )   (transposed for PV B-fragments)
// ---------------------------------------------------------------------------
__global__ void __launch_bounds__(32)
proj_kernel(const float* __restrict__ Qv, const float* __restrict__ Kv,
            const float* __restrict__ Vv, const unsigned short* __restrict__ wb,
            unsigned short* __restrict__ qp, unsigned short* __restrict__ kp,
            unsigned short* __restrict__ vt) {
  const int lane = threadIdx.x;
  const int half = lane >> 4;
  const int ln   = lane & 15;
  const int qt   = blockIdx.x % (SEQ / 16);
  const int rest = blockIdx.x / (SEQ / 16);
  const int b    = rest % BATCH;
  const int p    = rest / BATCH;                 // 0:q 1:k 2:v
  const float*          X  = (p == 0) ? Qv : (p == 1) ? Kv : Vv;
  const unsigned short* Wb = wb + p * (DDIM * EDIM);
  const int tok0 = b * SEQ + qt * 16;

  v8f O[4] = {};
#pragma unroll
  for (int c = 0; c < 8; ++c) {
    // A fragment: two contiguous runs of 8 f32 -> 4 x b128 + packed cvt
    const v4f* x0 = (const v4f*)(X + (tok0 + ln) * EDIM + c * 32 + 16 * half);
    const v4f* x1 = (const v4f*)(X + (tok0 + ln) * EDIM + c * 32 + 16 * half + 8);
    v4f a0 = x0[0], a1 = x0[1], a2 = x1[0], a3 = x1[1];
    Frag ax;
#pragma unroll
    for (int e = 0; e < 4; ++e) {
      ax.b[e]      = (__bf16)a0[e];
      ax.b[4 + e]  = (__bf16)a1[e];
      ax.b[8 + e]  = (__bf16)a2[e];
      ax.b[12 + e] = (__bf16)a3[e];
    }
#pragma unroll
    for (int nt = 0; nt < 4; ++nt) {
      const v4u* wr = (const v4u*)(Wb + (nt * 16 + ln) * EDIM + c * 32 + 16 * half);
      Frag bw;
      bw.q[0] = wr[0]; bw.q[1] = wr[1];
      O[nt] = __builtin_amdgcn_wmma_f32_16x16x32_bf16(false, ax.v, false, bw.v,
                                                      (short)0, O[nt], false, false);
    }
  }
  const float qscale = 0.125f * 1.44269504088896340736f;  // 1/sqrt(D) * log2(e)
#pragma unroll
  for (int nt = 0; nt < 4; ++nt)
#pragma unroll
    for (int r = 0; r < 8; ++r) {
      int n   = qt * 16 + r + 8 * half;
      int col = nt * 16 + ln;
      float o = O[nt][r];
      if (p == 0)      qp[(b * SEQ + n) * DDIM + col] = f2bf(o * qscale);
      else if (p == 1) kp[(b * SEQ + n) * DDIM + col] = f2bf(o);
      else             vt[(b * DDIM + col) * SEQ + n] = f2bf(o);
    }
}

// ---------------------------------------------------------------------------
// Flash attention without max-tracking (scores bounded, log2 domain):
//   P = exp2(S'), masked entries = exp2(-44) (equal -> exact uniform for dead rows)
//   l accumulated as a 5th WMMA tile against a ones B-matrix.
// ---------------------------------------------------------------------------
__global__ void __launch_bounds__(32)
attn_kernel(const unsigned short* __restrict__ qp,
            const unsigned short* __restrict__ kp,
            const unsigned short* __restrict__ vt,
            const unsigned char*  __restrict__ maskp,
            float* __restrict__ Opart, float* __restrict__ lpart) {
  __shared__ __bf16 PlB[32 * 16];                // [col][row], transposed staging

  const int lane = threadIdx.x;
  const int half = lane >> 4;
  const int ln   = lane & 15;

  const int qt   = blockIdx.x % (SEQ / 16);
  const int rest = blockIdx.x / (SEQ / 16);
  const int b    = rest % BATCH;
  const int sp   = rest / BATCH;                 // k-split index
  const int qbase  = qt * 16;
  const int jstart = sp * (SEQ / KSPLIT);
  const int jdiag  = qbase & ~31;                // the one chunk containing i==j

  // Q A-fragments: contiguous 16B runs, k = 16*half + e
  Frag aq[2];
#pragma unroll
  for (int c = 0; c < 2; ++c) {
    const v4u* q0 = (const v4u*)(qp + (b * SEQ + qbase + ln) * DDIM + c * 32 + 16 * half);
    const v4u* q1 = (const v4u*)(qp + (b * SEQ + qbase + ln) * DDIM + c * 32 + 16 * half + 8);
    aq[c].q[0] = q0[0];
    aq[c].q[1] = q1[0];
  }

  // ones B-fragment for the row-sum tile
  Frag ones;
#pragma unroll
  for (int e = 0; e < 16; ++e) ones.u[e] = 0x3F80;  // bf16 1.0

  bool rowm[8]; int irow[8];
#pragma unroll
  for (int r = 0; r < 8; ++r) {
    irow[r] = qbase + r + 8 * half;
    rowm[r] = maskp[b * SEQ + irow[r]] != 0;
  }

  v8f O[5] = {};                                 // O[4] = row-sum accumulator

  for (int jb = jstart; jb < jstart + SEQ / KSPLIT; jb += 32) {
    // ---- S' = (q*scale*log2e) . k, two 16-key tiles --------------------
    v8f S[2];
#pragma unroll
    for (int s = 0; s < 2; ++s) {
      Frag bk0, bk1;
      const v4u* k0 = (const v4u*)(kp + (b * SEQ + jb + s * 16 + ln) * DDIM + half * 16);
      const v4u* k1 = (const v4u*)(kp + (b * SEQ + jb + s * 16 + ln) * DDIM + 32 + half * 16);
      bk0.q[0] = k0[0]; bk0.q[1] = k0[1];
      bk1.q[0] = k1[0]; bk1.q[1] = k1[1];
      v8f acc = {};
      acc = __builtin_amdgcn_wmma_f32_16x16x32_bf16(false, aq[0].v, false, bk0.v,
                                                    (short)0, acc, false, false);
      acc = __builtin_amdgcn_wmma_f32_16x16x32_bf16(false, aq[1].v, false, bk1.v,
                                                    (short)0, acc, false, false);
      S[s] = acc;
    }
    // ---- mask (finite log2-domain sentinel) ---------------------------
#pragma unroll
    for (int s = 0; s < 2; ++s) {
      int  jcol = jb + s * 16 + ln;
      bool cm   = maskp[b * SEQ + jcol] != 0;
#pragma unroll
      for (int r = 0; r < 8; ++r) {
        bool keep = rowm[r] && cm;
        S[s][r] = keep ? S[s][r] : -44.0f;
      }
    }
    if (jb == jdiag) {                           // uniform branch, runs once
#pragma unroll
      for (int s = 0; s < 2; ++s) {
        int jcol = jb + s * 16 + ln;
#pragma unroll
        for (int r = 0; r < 8; ++r)
          if (irow[r] == jcol) S[s][r] = -44.0f;
      }
    }
    // ---- P = exp2(S'), stage to LDS as bf16 (packed b128 stores) ------
#pragma unroll
    for (int s = 0; s < 2; ++s) {
      union { __bf16 h[8]; v4u q; } pk;
#pragma unroll
      for (int r = 0; r < 8; ++r)
        pk.h[r] = (__bf16)__builtin_amdgcn_exp2f(S[s][r]);
      *(v4u*)&PlB[(s * 16 + ln) * 16 + 8 * half] = pk.q;
    }
    // single wave per block: DS ops are in-order within a wave
    Frag ap;
#pragma unroll
    for (int e = 0; e < 16; ++e)
      ap.b[e] = PlB[(e + 16 * half) * 16 + ln];
    // ---- O += P*V ; l += P*1 ------------------------------------------
#pragma unroll
    for (int nt = 0; nt < 4; ++nt) {
      Frag bv;
      const v4u* vp = (const v4u*)(vt + (b * DDIM + nt * 16 + ln) * SEQ + jb + half * 16);
      bv.q[0] = vp[0]; bv.q[1] = vp[1];
      O[nt] = __builtin_amdgcn_wmma_f32_16x16x32_bf16(false, ap.v, false, bv.v,
                                                      (short)0, O[nt], false, false);
    }
    O[4] = __builtin_amdgcn_wmma_f32_16x16x32_bf16(false, ap.v, false, ones.v,
                                                   (short)0, O[4], false, false);
  }

  // ---- store unnormalized partials -----------------------------------
  const int pb = (sp * BATCH + b) * SEQ;
  if (ln == 0) {
#pragma unroll
    for (int r = 0; r < 8; ++r)
      lpart[pb + qbase + r + 8 * half] = O[4][r];
  }
#pragma unroll
  for (int nt = 0; nt < 4; ++nt)
#pragma unroll
    for (int r = 0; r < 8; ++r)
      Opart[(pb + qbase + r + 8 * half) * DDIM + nt * 16 + ln] = O[nt][r];
}

// ---------------------------------------------------------------------------
// Combine split-K partials: out = sum_s O_s / sum_s l_s
// ---------------------------------------------------------------------------
__global__ void __launch_bounds__(256)
combine_kernel(const float* __restrict__ Opart, const float* __restrict__ lpart,
               float* __restrict__ out) {
  const int idx = blockIdx.x * 256 + threadIdx.x;    // (b*SEQ+n)*DDIM + d
  const int d  = idx % DDIM;
  const int bn = idx / DDIM;
  float lsum = 0.f, osum = 0.f;
#pragma unroll
  for (int s = 0; s < KSPLIT; ++s) {
    lsum += lpart[s * (BATCH * SEQ) + bn];
    osum += Opart[(size_t)(s * (BATCH * SEQ) + bn) * DDIM + d];
  }
  out[idx] = osum / lsum;
}

extern "C" void kernel_launch(void* const* d_in, const int* in_sizes, int n_in,
                              void* d_out, int out_size, void* d_ws, size_t ws_size,
                              hipStream_t stream) {
  const float*         Qv   = (const float*)d_in[0];
  const float*         Kv   = (const float*)d_in[1];
  const float*         Vv   = (const float*)d_in[2];
  const unsigned char* mask = (const unsigned char*)d_in[3];
  const float*         Wq   = (const float*)d_in[4];
  const float*         Wk   = (const float*)d_in[5];
  const float*         Wv   = (const float*)d_in[6];
  float*               out  = (float*)d_out;

  char* ws = (char*)d_ws;
  const size_t nBND = (size_t)BATCH * SEQ * DDIM;            // 1,048,576
  unsigned short* qp = (unsigned short*)(ws);
  unsigned short* kp = (unsigned short*)(ws + 2 * nBND);
  unsigned short* vt = (unsigned short*)(ws + 4 * nBND);
  float*          Op = (float*)(ws + 6 * nBND);              // [KSPLIT][B][N][D] f32
  float*          lp = (float*)(ws + 6 * nBND + 4 * KSPLIT * nBND);  // [KSPLIT][B][N]
  unsigned short* wb = (unsigned short*)(ws + 6 * nBND + 4 * KSPLIT * nBND
                                         + 4 * (size_t)KSPLIT * BATCH * SEQ);

  wcvt_kernel<<<(3 * DDIM * EDIM / 2) / 256, 256, 0, stream>>>(Wq, Wk, Wv,
                                                               (unsigned int*)wb);
  proj_kernel<<<3 * BATCH * (SEQ / 16), 32, 0, stream>>>(Qv, Kv, Vv, wb,
                                                         qp, kp, vt);
  attn_kernel<<<KSPLIT * BATCH * (SEQ / 16), 32, 0, stream>>>(qp, kp, vt, mask,
                                                              Op, lp);
  combine_kernel<<<(BATCH * SEQ * DDIM) / 256, 256, 0, stream>>>(Op, lp, out);
}
